// LSTM2_14637248545493
// MI455X (gfx1250) — matrix operand order repaired and checked by reading, not compiled, not run on hardware
//
#include <hip/hip_runtime.h>

typedef __attribute__((ext_vector_type(16))) _Float16 v16h;
typedef __attribute__((ext_vector_type(8)))  float    v8f;

#define TLEN 512
#define ILEN 6
#define HLEN 6
#define BATCH 8192

__device__ __forceinline__ float sigm(float x) {
    return __builtin_amdgcn_rcpf(1.0f + __expf(-x));
}
__device__ __forceinline__ float tanh_f(float x) {
    return 2.0f * __builtin_amdgcn_rcpf(1.0f + __expf(-2.0f * x)) - 1.0f;
}

// One wave handles a 16-sequence batch tile. Recurrent state lives in
// registers (c, f32) + a 256B per-wave LDS transpose buffer (h, f16).
// Per step: 3x global b64 (x), 1x ds load (h), 4x v_wmma_f32_16x16x32_f16
// (one per gate, bias pre-loaded into the C accumulator), lane-local
// activations, 8x ds_store_b16 (h write-back in A-fragment-friendly layout).
__global__ void __launch_bounds__(256)
lstm2_wmma_kernel(const float* __restrict__ x,
                  const float* __restrict__ W_ih, const float* __restrict__ W_hh,
                  const float* __restrict__ b_ih, const float* __restrict__ b_hh,
                  const float* __restrict__ W1,   const float* __restrict__ b1,
                  const float* __restrict__ W2,   const float* __restrict__ b2,
                  float* __restrict__ out)
{
    __shared__ __align__(16) char smem[8 * 512];   // 512 B per wave
    const int lane = threadIdx.x & 31;
    const int wid  = threadIdx.x >> 5;
    char* hb = smem + wid * 512;

    const int tile = blockIdx.x * 8 + wid;         // 512 tiles total
    const int b0   = tile * 16;
    const int n    = lane & 15;                    // column (hidden unit) in tile
    const int hi   = lane >> 4;                    // lane half

    // ---- constant B fragments (fused [W_ih|W_hh]^T, one 16x16 tile per gate,
    //      gate order i,f,g,o) and per-lane fused bias ----------------------
    // B layout (16x16x32 f16): lane = N, element e -> K = (e<8) ? 8*(lane>>4)+e : 16+8*(lane>>4)+(e-8)
    v16h Bfrag[4];
    float bias[4];
    #pragma unroll
    for (int q = 0; q < 4; ++q) {
        #pragma unroll
        for (int e = 0; e < 16; ++e) {
            int k = (e < 8) ? (8 * hi + e) : (16 + 8 * hi + (e - 8));
            float w = 0.0f;
            if (n < HLEN && k < ILEN + HLEN) {
                w = (k < ILEN) ? W_ih[(q * HLEN + n) * ILEN + k]
                               : W_hh[(q * HLEN + n) * HLEN + (k - ILEN)];
            }
            Bfrag[q][e] = (_Float16)w;
        }
        bias[q] = (n < HLEN) ? (b_ih[q * HLEN + n] + b_hh[q * HLEN + n]) : 0.0f;
    }

    // per-lane x row pointer (used by lanes 0-15; n == lane there)
    const float* xrow = x + (size_t)(b0 + n) * (TLEN * ILEN);

    // zero the h transpose buffer (h0 = 0)
    if (lane < 16) {
        *(uint4*)(hb + lane * 16) = make_uint4(0u, 0u, 0u, 0u);
    }
    asm volatile("s_wait_dscnt 0x0" ::: "memory");

    float c[8], h32[8];
    #pragma unroll
    for (int r = 0; r < 8; ++r) { c[r] = 0.0f; h32[r] = 0.0f; }

    for (int t = 0; t < TLEN; ++t) {
        // ---- assemble A fragment (16x32 f16, K = [x0..x5, h0..h5, 0...]) ----
        // lanes 0-15 : elems 0-7 carry K=0..7  -> x0..x5, h0, h1
        // lanes 16-31: elems 0-3 carry K=8..11 -> h2..h5
        v16h a;
        #pragma unroll
        for (int e = 0; e < 16; ++e) a[e] = (_Float16)0.0f;

        if (lane < 16) {
            const float2* xp = (const float2*)(xrow + t * ILEN);
            float2 x01 = xp[0], x23 = xp[1], x45 = xp[2];
            __builtin_prefetch(xrow + (t + 24) * ILEN, 0, 1);
            a[0] = (_Float16)x01.x; a[1] = (_Float16)x01.y;
            a[2] = (_Float16)x23.x; a[3] = (_Float16)x23.y;
            a[4] = (_Float16)x45.x; a[5] = (_Float16)x45.y;
            const _Float16* hp = (const _Float16*)(hb + lane * 16 + 8); // h0,h1
            a[6] = hp[0]; a[7] = hp[1];
        } else {
            const _Float16* hp = (const _Float16*)(hb + (lane - 16) * 16); // h2..h5
            a[0] = hp[0]; a[1] = hp[1]; a[2] = hp[2]; a[3] = hp[3];
        }

        // ---- gates = A x W_gate + bias : 4 WMMAs, bias via C operand ----
        v8f ci, cf, cg, co;
        #pragma unroll
        for (int r = 0; r < 8; ++r) {
            ci[r] = bias[0]; cf[r] = bias[1]; cg[r] = bias[2]; co[r] = bias[3];
        }
        v8f g0 = __builtin_amdgcn_wmma_f32_16x16x32_f16(false, a, false, Bfrag[0], (short)0, ci, false, false);
        v8f g1 = __builtin_amdgcn_wmma_f32_16x16x32_f16(false, a, false, Bfrag[1], (short)0, cf, false, false);
        v8f g2 = __builtin_amdgcn_wmma_f32_16x16x32_f16(false, a, false, Bfrag[2], (short)0, cg, false, false);
        v8f g3 = __builtin_amdgcn_wmma_f32_16x16x32_f16(false, a, false, Bfrag[3], (short)0, co, false, false);

        // ---- lane-local LSTM cell update (f32) ----
        #pragma unroll
        for (int r = 0; r < 8; ++r) {
            float ig = sigm(g0[r]);
            float fg = sigm(g1[r]);
            float gg = tanh_f(g2[r]);
            float og = sigm(g3[r]);
            float cn = fg * c[r] + ig * gg;
            c[r]   = cn;
            h32[r] = og * tanh_f(cn);
        }

        // ---- write h back as f16 in A-fragment-friendly transposed layout:
        //      row m (16B pitch): bytes 0-7 = h2..h5, bytes 8-11 = h0,h1 ----
        if (n < HLEN) {
            int off = (n < 2) ? (8 + 2 * n) : (2 * (n - 2));
            #pragma unroll
            for (int r = 0; r < 8; ++r) {
                int m = r + 8 * hi;
                *(_Float16*)(hb + m * 16 + off) = (_Float16)h32[r];
            }
        }
        asm volatile("s_wait_dscnt 0x0" ::: "memory");
    }

    // ---- MLP head + 2-way softmax (once per sequence) ----
    // stash final h (f32) row-major [16][6] in the same per-wave LDS region
    if (n < HLEN) {
        #pragma unroll
        for (int r = 0; r < 8; ++r) {
            int m = r + 8 * hi;
            *(float*)(hb + m * 24 + n * 4) = h32[r];
        }
    }
    asm volatile("s_wait_dscnt 0x0" ::: "memory");

    if (lane < 16) {
        const float* hv = (const float*)(hb + lane * 24);
        float z1[4];
        #pragma unroll
        for (int q = 0; q < 4; ++q) {
            float s = b1[q];
            #pragma unroll
            for (int u = 0; u < HLEN; ++u) s += W1[q * HLEN + u] * hv[u];
            z1[q] = s;
        }
        float z2[2];
        #pragma unroll
        for (int p = 0; p < 2; ++p) {
            float s = b2[p];
            #pragma unroll
            for (int q = 0; q < 4; ++q) s += W2[p * 4 + q] * z1[q];
            z2[p] = s;
        }
        float mx = fmaxf(z2[0], z2[1]);
        float e0 = __expf(z2[0] - mx);
        float e1 = __expf(z2[1] - mx);
        float inv = 1.0f / (e0 + e1);
        *(float2*)(out + (size_t)(b0 + lane) * 2) = make_float2(e0 * inv, e1 * inv);
    }
}

extern "C" void kernel_launch(void* const* d_in, const int* in_sizes, int n_in,
                              void* d_out, int out_size, void* d_ws, size_t ws_size,
                              hipStream_t stream) {
    (void)in_sizes; (void)n_in; (void)out_size; (void)d_ws; (void)ws_size;
    const float* x    = (const float*)d_in[0];
    const float* W_ih = (const float*)d_in[1];
    const float* W_hh = (const float*)d_in[2];
    const float* b_ih = (const float*)d_in[3];
    const float* b_hh = (const float*)d_in[4];
    const float* W1   = (const float*)d_in[5];
    const float* b1   = (const float*)d_in[6];
    const float* W2   = (const float*)d_in[7];
    const float* b2   = (const float*)d_in[8];
    float* out = (float*)d_out;

    // 8192 sequences / 16 per wave = 512 waves = 64 blocks x 8 waves
    dim3 grid(BATCH / (16 * 8));
    dim3 block(256);
    lstm2_wmma_kernel<<<grid, block, 0, stream>>>(x, W_ih, W_hh, b_ih, b_hh,
                                                  W1, b1, W2, b2, out);
}
